// GroupedQueryAttention_29008209117982
// MI455X (gfx1250) — matrix operand order repaired and checked by reading, not compiled, not run on hardware
//
#include <hip/hip_runtime.h>
#include <hip/hip_bf16.h>

typedef __attribute__((ext_vector_type(16))) __bf16 v16bf;
typedef __attribute__((ext_vector_type(8)))  __bf16 v8bf;
typedef __attribute__((ext_vector_type(8)))  float  v8f;

#define DIMC 2048
#define NH   16
#define KVH  4
#define HD   128
#define BB   2
#define TT   2048

// ---------------------------------------------------------------------------
// WMMA helpers (CDNA5 gfx1250, wave32). D = A(16x32) * B(32x16) + C, bf16->f32
// ---------------------------------------------------------------------------
static __device__ __forceinline__ v8f wmma_bf16(v16bf a, v16bf b, v8f c) {
    return __builtin_amdgcn_wmma_f32_16x16x32_bf16(
        /*neg_a=*/false, a, /*neg_b=*/false, b,
        /*c_mod=*/(short)0, c, /*reuse_a=*/false, /*reuse_b=*/false);
}

// A-frag: 16(M) x 32(K), row-major memory [M, K] with leading dim ld.
// ISA layout: lanes 0-15 -> M=lane, K=0..7 (v0-3) and 16..23 (v4-7);
//             lanes 16-31 -> M=lane-16, K=8..15 and 24..31.
static __device__ __forceinline__ v16bf load_frag_a(const __bf16* p0, int ld, int lane) {
    int m  = lane & 15;
    int kh = (lane >> 4) << 3;            // 0 or 8
    const __bf16* p = p0 + (size_t)m * ld + kh;
    v8bf lo = *(const v8bf*)(p);          // K = kh..kh+7
    v8bf hi = *(const v8bf*)(p + 16);     // K = 16+kh..23+kh
    return __builtin_shufflevector(lo, hi, 0,1,2,3,4,5,6,7,8,9,10,11,12,13,14,15);
}

// B-frag: 32(K) x 16(N) where B[k,n] is sourced from row-major memory [N, K]
// (i.e. the "NT" operand): lane n holds B_row n, K in the same order as A-frag.
// ISA layout: lanes 0-15 -> K=0..7 and 16..23; lanes 16-31 -> K=8..15 and 24..31; col N = lane&15.
static __device__ __forceinline__ v16bf load_frag_b(const __bf16* p0, int ld, int lane) {
    int n  = lane & 15;
    int kh = (lane >> 4) << 3;            // 0 or 8
    const __bf16* p = p0 + (size_t)n * ld + kh;
    v8bf lo = *(const v8bf*)(p);
    v8bf hi = *(const v8bf*)(p + 16);
    return __builtin_shufflevector(lo, hi, 0,1,2,3,4,5,6,7,8,9,10,11,12,13,14,15);
}

// ---------------------------------------------------------------------------
// f32 -> bf16 conversion
// ---------------------------------------------------------------------------
__global__ void cvt_f32_bf16(const float* __restrict__ in, __bf16* __restrict__ out, long n) {
    long i = (long)blockIdx.x * blockDim.x + threadIdx.x;
    long stride = (long)gridDim.x * blockDim.x;
    for (; i < n; i += stride) out[i] = (__bf16)in[i];
}

// ---------------------------------------------------------------------------
// NT-GEMM: C[m,n] = sum_k A[m,k] * B[n,k] (+ bias[n]); A,B bf16, C f32.
// Wave tile 64x64 (16 accum tiles), WG = 4 waves -> 128x128 per block.
// ---------------------------------------------------------------------------
__global__ __launch_bounds__(128)
void gemm_nt_bf16(const __bf16* __restrict__ A, const __bf16* __restrict__ Bm,
                  float* __restrict__ C, const float* __restrict__ bias,
                  int M, int N, int K) {
    int lane = threadIdx.x & 31;
    int wid  = threadIdx.x >> 5;
    int m0 = blockIdx.y * 128 + (wid >> 1) * 64;
    int n0 = blockIdx.x * 128 + (wid & 1) * 64;

    v8f acc[4][4] = {};

    for (int k0 = 0; k0 < K; k0 += 32) {
        if (k0 + 32 < K) {  // prefetch next K-slab (global_prefetch_b8)
            __builtin_prefetch((const void*)(A + (size_t)(m0 + (lane << 1)) * K + k0 + 32), 0, 1);
            __builtin_prefetch((const void*)(Bm + (size_t)(n0 + (lane << 1)) * K + k0 + 32), 0, 1);
        }
        v16bf a[4], b[4];
        #pragma unroll
        for (int i = 0; i < 4; i++)
            a[i] = load_frag_a(A + (size_t)(m0 + 16 * i) * K + k0, K, lane);
        #pragma unroll
        for (int j = 0; j < 4; j++)
            b[j] = load_frag_b(Bm + (size_t)(n0 + 16 * j) * K + k0, K, lane);
        #pragma unroll
        for (int i = 0; i < 4; i++)
            #pragma unroll
            for (int j = 0; j < 4; j++)
                acc[i][j] = wmma_bf16(a[i], b[j], acc[i][j]);
    }

    int nl = lane & 15;
    int mh = (lane >> 4) << 3;
    #pragma unroll
    for (int i = 0; i < 4; i++)
        #pragma unroll
        for (int j = 0; j < 4; j++) {
            int cn = n0 + 16 * j + nl;
            float bv = bias ? bias[cn] : 0.0f;
            float* cp = C + (size_t)(m0 + 16 * i + mh) * N + cn;
            #pragma unroll
            for (int r = 0; r < 8; r++) cp[(size_t)r * N] = acc[i][j][r] + bv;
        }
}

// ---------------------------------------------------------------------------
// RoPE (reference variant: out = rot * (sin + cos)) + reshape to [B,h,T,HD] bf16
// ---------------------------------------------------------------------------
__global__ void rope_to_heads(const float* __restrict__ proj, const float* __restrict__ cosb,
                              const float* __restrict__ sinb, __bf16* __restrict__ out, int nh) {
    long total = (long)BB * TT * nh * HD;
    long i = (long)blockIdx.x * blockDim.x + threadIdx.x;
    long stride = (long)gridDim.x * blockDim.x;
    for (; i < total; i += stride) {
        int  d  = (int)(i & (HD - 1));
        long r  = i >> 7;
        int  t  = (int)(r & (TT - 1));
        long r2 = r >> 11;
        int  h  = (int)(r2 % nh);
        int  b  = (int)(r2 / nh);
        const float* row = proj + ((long)(b * TT + t) * nh + h) * HD;
        float rot = (d & 1) ? row[d - 1] : -row[d + 1];
        float cs  = cosb[t * HD + d] + sinb[t * HD + d];
        out[i] = (__bf16)(rot * cs);
    }
}

// V: [ (b*T+t), kvh*HD ] f32 -> Vt[ b, kvh, d, t ] bf16 (transposed per head)
__global__ void v_transpose(const float* __restrict__ vproj, __bf16* __restrict__ vt) {
    long total = (long)BB * KVH * HD * TT;
    long i = (long)blockIdx.x * blockDim.x + threadIdx.x;
    long stride = (long)gridDim.x * blockDim.x;
    for (; i < total; i += stride) {
        int  t  = (int)(i & (TT - 1));
        long r  = i >> 11;
        int  d  = (int)(r & (HD - 1));
        long r2 = r >> 7;
        int  kv = (int)(r2 & (KVH - 1));
        int  b  = (int)(r2 >> 2);
        vt[i] = (__bf16)vproj[(long)(b * TT + t) * (KVH * HD) + kv * HD + d];
    }
}

// ---------------------------------------------------------------------------
// Flash attention, one wave = 16 query rows of one (b,h).
// Computes S^T = K*Q^T so that P tiles are produced directly in A-frag layout.
// ---------------------------------------------------------------------------
__global__ __launch_bounds__(128)
void attn_kernel(const __bf16* __restrict__ Q, const __bf16* __restrict__ Kc,
                 const __bf16* __restrict__ Vt, __bf16* __restrict__ Oatt) {
    int lane = threadIdx.x & 31;
    int wid  = threadIdx.x >> 5;
    int bh = blockIdx.y;                 // b*NH + h
    int b  = bh >> 4, h = bh & 15;
    int kv = h >> 2;                     // GQA: 4 Q heads per KV head
    int q0 = blockIdx.x * 64 + wid * 16;

    const __bf16* Qh = Q  + (size_t)bh * TT * HD;
    const __bf16* Kh = Kc + (size_t)(b * KVH + kv) * TT * HD;
    const __bf16* Vh = Vt + (size_t)(b * KVH + kv) * HD * TT;

    // Q as B-operand (d x m), loaded once: 4 chunks over head_dim
    v16bf qb[4];
    #pragma unroll
    for (int c = 0; c < 4; c++)
        qb[c] = load_frag_b(Qh + (size_t)q0 * HD + 32 * c, HD, lane);

    v8f o[8] = {};
    float mrun = -1e30f, lrun = 0.0f;
    const float scale = 0.08838834764831845f;  // 1/sqrt(128)
    int qcol = q0 + (lane & 15);               // query handled by this lane's stats
    int mh   = (lane >> 4) << 3;

    for (int s0 = 0; s0 < q0 + 16; s0 += 32) {
        v8f slo = {}, shi = {};
        #pragma unroll
        for (int c = 0; c < 4; c++) {
            v16bf ka = load_frag_a(Kh + (size_t)s0 * HD + 32 * c, HD, lane);
            slo = wmma_bf16(ka, qb[c], slo);        // S^T rows s0..s0+15
        }
        #pragma unroll
        for (int c = 0; c < 4; c++) {
            v16bf ka = load_frag_a(Kh + (size_t)(s0 + 16) * HD + 32 * c, HD, lane);
            shi = wmma_bf16(ka, qb[c], shi);        // S^T rows s0+16..s0+31
        }
        // scale + causal mask; S^T element (s = s0+mh+r [+16], q = qcol)
        float mx = -1e30f;
        #pragma unroll
        for (int r = 0; r < 8; r++) {
            int sl = s0 + mh + r;
            float a  = (sl      <= qcol) ? slo[r] * scale : -1e9f;
            float c2 = (sl + 16 <= qcol) ? shi[r] * scale : -1e9f;
            slo[r] = a; shi[r] = c2;
            mx = fmaxf(mx, fmaxf(a, c2));
        }
        mx = fmaxf(mx, __shfl_xor(mx, 16, 32));     // other half of this query's keys
        float mnew  = fmaxf(mrun, mx);
        float alpha = __expf(mrun - mnew);
        mrun = mnew;

        float rs = 0.0f;
        v16bf pa;                                    // P already in A-frag layout!
        #pragma unroll
        for (int r = 0; r < 8; r++) {
            float pl = __expf(slo[r] - mnew);
            float ph = __expf(shi[r] - mnew);
            rs += pl + ph;
            pa[r]     = (__bf16)pl;
            pa[8 + r] = (__bf16)ph;
        }
        rs += __shfl_xor(rs, 16, 32);
        lrun = lrun * alpha + rs;

        // broadcast alpha by output row (O is in C-layout: row m = mh+r)
        float af[8];
        #pragma unroll
        for (int r = 0; r < 8; r++) af[r] = __shfl(alpha, mh + r, 32);

        #pragma unroll
        for (int t = 0; t < 8; t++) {
            #pragma unroll
            for (int r = 0; r < 8; r++) o[t][r] *= af[r];
            v16bf vb = load_frag_b(Vh + (size_t)(t * 16) * TT + s0, TT, lane);
            o[t] = wmma_bf16(pa, vb, o[t]);
        }
    }

    float lf[8];
    #pragma unroll
    for (int r = 0; r < 8; r++) lf[r] = 1.0f / __shfl(lrun, mh + r, 32);

    int nl = lane & 15;
    #pragma unroll
    for (int t = 0; t < 8; t++)
        #pragma unroll
        for (int r = 0; r < 8; r++) {
            long row = (long)b * TT + q0 + mh + r;
            Oatt[row * DIMC + h * HD + t * 16 + nl] = (__bf16)(o[t][r] * lf[r]);
        }
}

// ---------------------------------------------------------------------------
extern "C" void kernel_launch(void* const* d_in, const int* in_sizes, int n_in,
                              void* d_out, int out_size, void* d_ws, size_t ws_size,
                              hipStream_t stream) {
    const float* x    = (const float*)d_in[0];
    const float* cosb = (const float*)d_in[1];
    const float* sinb = (const float*)d_in[2];
    // d_in[3] = bool mask (causal; computed analytically in-kernel)
    const float* Wq   = (const float*)d_in[4];
    const float* Wk   = (const float*)d_in[5];
    const float* Wv   = (const float*)d_in[6];
    const float* Wo   = (const float*)d_in[7];
    const float* bo   = (const float*)d_in[8];
    float* out = (float*)d_out;

    char* p = (char*)d_ws;
    auto take = [&](size_t bytes) -> char* {
        char* r = p;
        p += (bytes + 255) & ~(size_t)255;
        return r;
    };
    const long MROWS = (long)BB * TT;                 // 4096
    __bf16* xbf   = (__bf16*)take(MROWS * DIMC * 2);
    __bf16* Wqbf  = (__bf16*)take((size_t)DIMC * DIMC * 2);
    __bf16* Wkbf  = (__bf16*)take((size_t)KVH * HD * DIMC * 2);
    __bf16* Wvbf  = (__bf16*)take((size_t)KVH * HD * DIMC * 2);
    __bf16* Wobf  = (__bf16*)take((size_t)DIMC * DIMC * 2);
    float*  Qproj = (float*) take(MROWS * DIMC * 4);
    float*  Kproj = (float*) take(MROWS * KVH * HD * 4);
    float*  Vproj = (float*) take(MROWS * KVH * HD * 4);
    __bf16* Qr    = (__bf16*)take(MROWS * DIMC * 2);          // [B,NH,T,HD]
    __bf16* Kr    = (__bf16*)take(MROWS * KVH * HD * 2);      // [B,KVH,T,HD]
    __bf16* Vtb   = (__bf16*)take(MROWS * KVH * HD * 2);      // [B,KVH,HD,T]
    __bf16* Oatt  = (__bf16*)take(MROWS * DIMC * 2);          // [B*T, DIM]

    // 1) downcast activations + weights to bf16
    cvt_f32_bf16<<<1024, 256, 0, stream>>>(x,  xbf,  MROWS * DIMC);
    cvt_f32_bf16<<<1024, 256, 0, stream>>>(Wq, Wqbf, (long)DIMC * DIMC);
    cvt_f32_bf16<<<512,  256, 0, stream>>>(Wk, Wkbf, (long)KVH * HD * DIMC);
    cvt_f32_bf16<<<512,  256, 0, stream>>>(Wv, Wvbf, (long)KVH * HD * DIMC);
    cvt_f32_bf16<<<1024, 256, 0, stream>>>(Wo, Wobf, (long)DIMC * DIMC);

    // 2) projections (NT GEMMs: x @ W^T)
    gemm_nt_bf16<<<dim3(DIMC / 128, MROWS / 128), 128, 0, stream>>>(
        xbf, Wqbf, Qproj, nullptr, (int)MROWS, DIMC, DIMC);
    gemm_nt_bf16<<<dim3((KVH * HD) / 128, MROWS / 128), 128, 0, stream>>>(
        xbf, Wkbf, Kproj, nullptr, (int)MROWS, KVH * HD, DIMC);
    gemm_nt_bf16<<<dim3((KVH * HD) / 128, MROWS / 128), 128, 0, stream>>>(
        xbf, Wvbf, Vproj, nullptr, (int)MROWS, KVH * HD, DIMC);

    // 3) RoPE + head layout; V transposed per head
    rope_to_heads<<<1024, 256, 0, stream>>>(Qproj, cosb, sinb, Qr, NH);
    rope_to_heads<<<512,  256, 0, stream>>>(Kproj, cosb, sinb, Kr, KVH);
    v_transpose  <<<512,  256, 0, stream>>>(Vproj, Vtb);

    // 4) causal flash attention (WMMA)
    attn_kernel<<<dim3(TT / 64, BB * NH), 128, 0, stream>>>(Qr, Kr, Vtb, Oatt);

    // 5) output projection + bias
    gemm_nt_bf16<<<dim3(DIMC / 128, MROWS / 128), 128, 0, stream>>>(
        Oatt, Wobf, out, bo, (int)MROWS, DIMC, DIMC);
}
